// RepWalk_9242769621361
// MI455X (gfx1250) — matrix operand order repaired and checked by reading, not compiled, not run on hardware
//
#include <hip/hip_runtime.h>
#include <hip/hip_bf16.h>
#include <math.h>

// ---------------------------------------------------------------------------
// RepWalk forward for MI455X (gfx1250, wave32, WMMA).
// Big GEMMs: v_wmma_f32_16x16x32_f16, 128x64 block tile, 2x2 tiles per wave.
// A-tile staging uses GLOBAL_LOAD_ASYNC_TO_LDS_B128 (ASYNCcnt) -- CDNA5 async
// data mover path; completion via s_wait_asynccnt 0 + barrier.
// Sequential LSTM recurrence = per-step kernel launches (graph captured).
// ---------------------------------------------------------------------------

typedef __attribute__((ext_vector_type(16))) _Float16 v16h;
typedef __attribute__((ext_vector_type(8)))  float    v8f;
typedef __attribute__((ext_vector_type(4)))  _Float16 h4;

#define BN 64
#define SN 128
#define AN 8
#define HN 300
#define WDN 300
#define KP 320         // padded input/hidden K for LSTM GEMMs
#define HPAD 304       // per-gate padded hidden
#define GCOL 1216      // 4*HPAD gate-major columns (also concat K pad: 1200+16)
#define NPAD 640       // padded 2H (600) for v/e/lin1 buffers
#define NEGV (-1e9f)

// ---- CDNA5 async global->LDS staging (inline asm; bypasses builtin arity) --
// Low 32 bits of a flat pointer into LDS are the workgroup-relative LDS byte
// address.
__device__ __forceinline__ void async_g2l_b128(void* lds_ptr, const void* gptr) {
  unsigned int loff = (unsigned int)(unsigned long long)lds_ptr;
  asm volatile("global_load_async_to_lds_b128 %0, %1, off"
               :: "v"(loff), "v"((unsigned long long)gptr)
               : "memory");
}
__device__ __forceinline__ void wait_async0() {
  asm volatile("s_wait_asynccnt 0" ::: "memory");
}

// ---------------- WMMA fragment helpers (CDNA5 documented layouts) ----------
// A 16x32 f16: lanes0-15 M=0..15 {K=base..base+7, 16+base..16+base+7}, base=(lane>>4)*8
__device__ __forceinline__ v16h frag_A_lds(const _Float16* As, int mrow, int lane) {
  int base = (lane >> 4) * 8;
  const _Float16* p = As + mrow * 40;
  union { v16h v; h4 q[4]; } u;
  u.q[0] = *(const h4*)(p + base);
  u.q[1] = *(const h4*)(p + base + 4);
  u.q[2] = *(const h4*)(p + 16 + base);
  u.q[3] = *(const h4*)(p + 16 + base + 4);
  return u.v;
}
// B 32x16 f16 from LDS stored transposed [ncol][k]: lane n=lane&15, K={kbase..kbase+7, 16+kbase..16+kbase+7}, kbase=(lane>>4)*8
__device__ __forceinline__ v16h frag_B_ldsT(const _Float16* BsT, int ncol, int lane) {
  int kbase = (lane >> 4) * 8;
  const _Float16* p = BsT + ncol * 40 + kbase;
  union { v16h v; h4 q[4]; } u;
  u.q[0] = *(const h4*)(p);
  u.q[1] = *(const h4*)(p + 4);
  u.q[2] = *(const h4*)(p + 16);
  u.q[3] = *(const h4*)(p + 20);
  return u.v;
}

// ---------------- Generic batched WMMA GEMM: C[M,N] = A[M,K] * B[K,N] (+bias)
// block = 256 threads (8 waves); 128x64 output tile; each wave owns 2x2
// 16x16 sub-tiles (4 WMMAs / K-step). grid = (N/64, M/128, batch).
template<bool OUT16>
__global__ void gemm_wmma_kernel(const _Float16* __restrict__ A, int lda, long sA,
                                 const _Float16* __restrict__ Bm, int ldb, long sB,
                                 void* Cv, int ldc, long sC,
                                 int K, const float* __restrict__ bias) {
  __shared__ _Float16 As[128 * 40];   // A tile 128x32 (stride 40)
  __shared__ _Float16 BsT[64 * 40];   // B tile transposed [n][k]
  int bz = blockIdx.z;
  A  += (long)bz * sA;
  Bm += (long)bz * sB;
  int n0 = blockIdx.x * 64, m0 = blockIdx.y * 128;
  int tid = threadIdx.x, lane = tid & 31, wv = tid >> 5;
  int mt0 = (wv & 3) * 2;       // m sub-tile pair: tiles mt0, mt0+1 (of 8)
  int nt0 = (wv >> 2) * 2;      // n sub-tile pair: tiles nt0, nt0+1 (of 4)
  v8f acc00 = {}; v8f acc01 = {}; v8f acc10 = {}; v8f acc11 = {};
  for (int k0 = 0; k0 < K; k0 += 32) {
    // stage A 128x32 via async global->LDS B128 (2 chunks / thread)
    for (int c = tid; c < 512; c += 256) {
      int r = c >> 2, c8 = (c & 3) * 8;
      async_g2l_b128(&As[r * 40 + c8], &A[(long)(m0 + r) * lda + k0 + c8]);
    }
    // stage B 32x64 -> transposed LDS (VGPR transpose path)
    for (int c = tid; c < 512; c += 256) {
      int r = c >> 4, c4 = (c & 15) * 4;
      h4 t = *(const h4*)&Bm[(long)(k0 + r) * ldb + n0 + c4];
      BsT[(c4 + 0) * 40 + r] = t[0];
      BsT[(c4 + 1) * 40 + r] = t[1];
      BsT[(c4 + 2) * 40 + r] = t[2];
      BsT[(c4 + 3) * 40 + r] = t[3];
    }
    if (k0 + 32 < K)                           // prefetch next B tile
      __builtin_prefetch(&Bm[(long)(k0 + 32 + (tid >> 3)) * ldb + n0], 0, 1);
    wait_async0();
    __syncthreads();
    int lr = lane & 15;
    v16h a0 = frag_A_lds(As, (mt0 + 0) * 16 + lr, lane);
    v16h a1 = frag_A_lds(As, (mt0 + 1) * 16 + lr, lane);
    v16h b0 = frag_B_ldsT(BsT, (nt0 + 0) * 16 + lr, lane);
    v16h b1 = frag_B_ldsT(BsT, (nt0 + 1) * 16 + lr, lane);
    acc00 = __builtin_amdgcn_wmma_f32_16x16x32_f16(false, a0, false, b0, (short)0, acc00, false, false);
    acc01 = __builtin_amdgcn_wmma_f32_16x16x32_f16(false, a0, false, b1, (short)0, acc01, false, false);
    acc10 = __builtin_amdgcn_wmma_f32_16x16x32_f16(false, a1, false, b0, (short)0, acc10, false, false);
    acc11 = __builtin_amdgcn_wmma_f32_16x16x32_f16(false, a1, false, b1, (short)0, acc11, false, false);
    __syncthreads();
  }
  int ncol = lane & 15, mb = (lane >> 4) * 8;
  int gm0 = m0 + (mt0 + 0) * 16 + mb;
  int gm1 = m0 + (mt0 + 1) * 16 + mb;
  int gn0 = n0 + (nt0 + 0) * 16 + ncol;
  int gn1 = n0 + (nt0 + 1) * 16 + ncol;
  float bv0 = bias ? bias[gn0] : 0.f;
  float bv1 = bias ? bias[gn1] : 0.f;
#pragma unroll
  for (int r = 0; r < 8; ++r) {
    if (OUT16) {
      _Float16* C = (_Float16*)Cv + (long)bz * sC;
      C[(long)(gm0 + r) * ldc + gn0] = (_Float16)(acc00[r] + bv0);
      C[(long)(gm0 + r) * ldc + gn1] = (_Float16)(acc01[r] + bv1);
      C[(long)(gm1 + r) * ldc + gn0] = (_Float16)(acc10[r] + bv0);
      C[(long)(gm1 + r) * ldc + gn1] = (_Float16)(acc11[r] + bv1);
    } else {
      float* C = (float*)Cv + (long)bz * sC;
      C[(long)(gm0 + r) * ldc + gn0] = acc00[r] + bv0;
      C[(long)(gm0 + r) * ldc + gn1] = acc01[r] + bv1;
      C[(long)(gm1 + r) * ldc + gn0] = acc10[r] + bv0;
      C[(long)(gm1 + r) * ldc + gn1] = acc11[r] + bv1;
    }
  }
}

// ---------------- LSTM one time step (both directions), WMMA h@Whh^T + gates
// grid = (19 j-tiles, 2 directions). gpre holds x@Wih^T + bih + bhh (gate-major).
__global__ void lstm_step_kernel(const _Float16* __restrict__ gpre,   // [2][64*T][GCOL]
                                 const _Float16* __restrict__ WhhT,   // [2][KP][GCOL]
                                 const _Float16* __restrict__ h_read, // [2][64][KP]
                                 _Float16* __restrict__ h_write,      // [2][64][KP]
                                 float* __restrict__ cst,             // [2][64][HPAD]
                                 float* __restrict__ out,             // [64][T][NPAD]
                                 const int* __restrict__ lens,
                                 int t, int T) {
  __shared__ _Float16 As[64 * 40];
  __shared__ _Float16 BsT[64 * 40];
  __shared__ float gsm[64 * 68];
  int jt = blockIdx.x;           // hidden j tile: j in [jt*16, jt*16+16)
  int d  = blockIdx.y;           // direction
  const _Float16* Adat = h_read + (long)d * 64 * KP;
  const _Float16* W    = WhhT + (long)d * KP * GCOL;
  int tid = threadIdx.x, lane = tid & 31, wv = tid >> 5;
  int mt = wv & 3, nt0 = (wv >> 2) * 2;
  v8f acc0 = {}; v8f acc1 = {};
  for (int ks = 0; ks < KP / 32; ++ks) {
    int k0 = ks * 32;
    // stage A = h [64][32] via async global->LDS B128 (1 chunk / thread)
    for (int c = tid; c < 256; c += 256) {
      int r = c >> 2, c8 = (c & 3) * 8;
      async_g2l_b128(&As[r * 40 + c8], &Adat[(long)r * KP + k0 + c8]);
    }
    for (int c = tid; c < 512; c += 256) {        // stage B: 4 gate column groups
      int r = c >> 4, c4 = (c & 15) * 4;
      int g = c4 >> 4, jj = c4 & 15;
      h4 tv = *(const h4*)&W[(long)(k0 + r) * GCOL + g * HPAD + jt * 16 + jj];
      BsT[(c4 + 0) * 40 + r] = tv[0];
      BsT[(c4 + 1) * 40 + r] = tv[1];
      BsT[(c4 + 2) * 40 + r] = tv[2];
      BsT[(c4 + 3) * 40 + r] = tv[3];
    }
    wait_async0();
    __syncthreads();
    v16h af = frag_A_lds(As, mt * 16 + (lane & 15), lane);
    v16h b0 = frag_B_ldsT(BsT, nt0 * 16 + (lane & 15), lane);
    v16h b1 = frag_B_ldsT(BsT, (nt0 + 1) * 16 + (lane & 15), lane);
    acc0 = __builtin_amdgcn_wmma_f32_16x16x32_f16(false, af, false, b0, (short)0, acc0, false, false);
    acc1 = __builtin_amdgcn_wmma_f32_16x16x32_f16(false, af, false, b1, (short)0, acc1, false, false);
    __syncthreads();
  }
  {
    int ncol = lane & 15, mb = (lane >> 4) * 8;
#pragma unroll
    for (int r = 0; r < 8; ++r) {
      gsm[(mt * 16 + mb + r) * 68 + nt0 * 16 + ncol]       = acc0[r];
      gsm[(mt * 16 + mb + r) * 68 + (nt0 + 1) * 16 + ncol] = acc1[r];
    }
  }
  __syncthreads();
  const _Float16* gp = gpre + (long)d * 64 * T * GCOL;
  for (int p = tid; p < 1024; p += 256) {
    int b = p >> 4, jj = p & 15, j = jt * 16 + jj;
    if (j >= HN) continue;
    int L = lens[b];
    long hidx = (long)d * 64 * KP + (long)b * KP + j;
    if (t < L) {
      int tok = (d == 0) ? t : (L - 1 - t);
      long gi = ((long)(b * T + tok)) * GCOL;
      float vi = (float)gp[gi + 0 * HPAD + j] + gsm[b * 68 + 0 * 16 + jj];
      float vf = (float)gp[gi + 1 * HPAD + j] + gsm[b * 68 + 1 * 16 + jj];
      float vg = (float)gp[gi + 2 * HPAD + j] + gsm[b * 68 + 2 * 16 + jj];
      float vo = (float)gp[gi + 3 * HPAD + j] + gsm[b * 68 + 3 * 16 + jj];
      float si = 1.f / (1.f + __expf(-vi));
      float sf = 1.f / (1.f + __expf(-vf));
      float so = 1.f / (1.f + __expf(-vo));
      float tg = tanhf(vg);
      long ci = (long)d * 64 * HPAD + (long)b * HPAD + j;
      float cn = sf * cst[ci] + si * tg;
      float hn = so * tanhf(cn);
      cst[ci] = cn;
      int pos = (d == 0) ? t : (L - 1 - t);
      out[(long)b * T * NPAD + (long)pos * NPAD + d * HN + j] = hn;
      h_write[hidx] = (_Float16)hn;
    } else {
      h_write[hidx] = h_read[hidx];   // frozen state
    }
  }
}

// ---------------- small utility kernels ------------------------------------
__global__ void zero_kernel(unsigned int* p, long n) {
  long i = (long)blockIdx.x * 256 + threadIdx.x;
  if (i < n) p[i] = 0u;
}

__global__ void lens_kernel(const int* text, const int* asp, int* lens) {
  int b = threadIdx.x;
  if (b < BN) {
    int c = 0;
    for (int s = 0; s < SN; ++s) c += (text[b * SN + s] != 0);
    lens[b] = c;
    int c2 = 0;
    for (int a = 0; a < AN; ++a) c2 += (asp[b * AN + a] != 0);
    lens[BN + b] = c2;
  }
}

// W [2][1200][300] -> gate-major padded transposed f16 [2][KP][GCOL]
__global__ void conv_wT_kernel(const float* __restrict__ W, _Float16* __restrict__ dst) {
  long i = (long)blockIdx.x * 256 + threadIdx.x;
  if (i >= 2L * KP * GCOL) return;
  int col = (int)(i % GCOL);
  long r = i / GCOL;
  int k = (int)(r % KP), d = (int)(r / KP);
  int g = col / HPAD, j = col % HPAD;
  float val = 0.f;
  if (k < HN && j < HN) val = W[((long)d * 4 * HN + g * HN + j) * WDN + k];
  dst[i] = (_Float16)val;
}

__global__ void bsum_kernel(const float* bih, const float* bhh, float* bsum) {
  int i = blockIdx.x * 256 + threadIdx.x;
  if (i >= 2 * GCOL) return;
  int col = i % GCOL, d = i / GCOL, g = col / HPAD, j = col % HPAD;
  bsum[i] = (j < HN) ? (bih[d * 4 * HN + g * HN + j] + bhh[d * 4 * HN + g * HN + j]) : 0.f;
}

// lin1_W [2][600][1200] -> [2][GCOL=1216][NPAD=640] f16, zero padded
__global__ void conv_lin1_kernel(const float* __restrict__ W, _Float16* __restrict__ dst) {
  long i = (long)blockIdx.x * 256 + threadIdx.x;
  if (i >= 2L * GCOL * NPAD) return;
  int n = (int)(i % NPAD);
  long r = i / NPAD;
  int k = (int)(r % GCOL), d = (int)(r / GCOL);
  float v = 0.f;
  if (k < 2 * HN * 2 && n < 2 * HN) v = W[((long)d * 2 * HN + n) * (4 * HN) + k];
  dst[i] = (_Float16)v;
}

__global__ void pad_lin1b_kernel(const float* b, float* dst) {
  int i = blockIdx.x * 256 + threadIdx.x;
  if (i >= 2 * NPAD) return;
  int n = i % NPAD, d = i / NPAD;
  dst[i] = (n < 2 * HN) ? b[d * 2 * HN + n] : 0.f;
}

__global__ void gather_kernel(const int* __restrict__ ids, const float* __restrict__ emb,
                              _Float16* __restrict__ dst, long ntok) {
  long i = (long)blockIdx.x * 256 + threadIdx.x;
  if (i >= ntok * KP) return;
  int k = (int)(i % KP);
  long tok = i / KP;
  int id = ids[tok];
  dst[i] = (_Float16)((k < WDN) ? emb[(long)id * WDN + k] : 0.f);
}

__global__ void cvt_f32_f16_kernel(const float* __restrict__ src, _Float16* __restrict__ dst, long n) {
  long i = (long)blockIdx.x * 256 + threadIdx.x;
  if (i < n) dst[i] = (_Float16)src[i];
}

// e [B][8][NPAD] f32 -> e16 [B][64][NPAD] (rows>=8 zero) and eT16 [B][NPAD][64]
__global__ void build_e16_kernel(const float* __restrict__ e, _Float16* __restrict__ e16,
                                 _Float16* __restrict__ eT16) {
  long i = (long)blockIdx.x * 256 + threadIdx.x;
  if (i >= (long)BN * 64 * NPAD) return;
  int k = (int)(i % NPAD);
  long r = i / NPAD;
  int a = (int)(r % 64), b = (int)(r / 64);
  float val = (a < AN) ? e[((long)b * AN + a) * NPAD + k] : 0.f;
  e16[i] = (_Float16)val;
  eT16[((long)b * NPAD + k) * 64 + a] = (_Float16)val;
}

__global__ void attn_softmax_kernel(const float* __restrict__ a32, const int* __restrict__ text,
                                    const int* __restrict__ asp, _Float16* __restrict__ a16) {
  int i = blockIdx.x * 256 + threadIdx.x;
  if (i >= BN * SN) return;
  int b = i >> 7;
  bool ms = (text[i] != 0);
  float vals[AN];
  float mx = -1e30f;
#pragma unroll
  for (int q = 0; q < AN; ++q) {
    bool tm = (asp[b * AN + q] != 0);
    float x = (ms && tm) ? a32[(long)i * 64 + q] : NEGV;
    vals[q] = x;
    mx = fmaxf(mx, x);
  }
  float sum = 0.f;
#pragma unroll
  for (int q = 0; q < AN; ++q) { vals[q] = __expf(vals[q] - mx); sum += vals[q]; }
  float inv = 1.f / sum;
#pragma unroll
  for (int q = 0; q < AN; ++q) a16[(long)i * 64 + q] = (_Float16)(vals[q] * inv);
  for (int q = AN; q < 64; ++q) a16[(long)i * 64 + q] = (_Float16)0.f;
}

__global__ void concat_kernel(const float* __restrict__ mid, const float* __restrict__ v,
                              _Float16* __restrict__ cc) {
  long i = (long)blockIdx.x * 256 + threadIdx.x;
  if (i >= (long)BN * SN * GCOL) return;
  int k = (int)(i % GCOL);
  long r = i / GCOL;
  float val;
  if (k < 2 * HN)           val = mid[r * NPAD + k];
  else if (k < 4 * HN)      val = v[r * NPAD + (k - 2 * HN)];
  else                      val = 0.f;
  cc[i] = (_Float16)val;
}

__global__ void tgate_kernel(const float* __restrict__ v, const float* __restrict__ W2,
                             const float* __restrict__ b2, float* __restrict__ tg) {
  int r = blockIdx.x * 256 + threadIdx.x;
  if (r >= BN * SN) return;
  float s = b2[0];
  for (int k = 0; k < 2 * HN; ++k) s += v[(long)r * NPAD + k] * W2[k];
  tg[r] = 1.f / (1.f + __expf(-s));
}

__global__ void vupdate_kernel(float* __restrict__ v, _Float16* __restrict__ vf,
                               const float* __restrict__ lin1out, const float* __restrict__ tg,
                               const float* __restrict__ pw) {
  long i = (long)blockIdx.x * 256 + threadIdx.x;
  if (i >= (long)BN * SN * NPAD) return;
  int k = (int)(i % NPAD);
  if (k >= 2 * HN) return;   // pad cols stay zero
  long r = i / NPAD;
  float m = lin1out[i];
  m = (m > 0.f) ? m : 0.01f * m;                    // leaky_relu
  float t = tg[r];
  float nv = ((1.f - t) * m + t * v[i]) * pw[r];
  v[i] = nv;
  vf[i] = (_Float16)nv;
}

__global__ void query_kernel(const float* __restrict__ e, const int* __restrict__ asp,
                             float* __restrict__ query) {
  int i = blockIdx.x * 256 + threadIdx.x;
  if (i >= BN * 2 * HN) return;
  int b = i / (2 * HN), k = i % (2 * HN);
  float q = NEGV;
  for (int a = 0; a < AN; ++a) {
    float cand = (asp[b * AN + a] != 0) ? e[((long)b * AN + a) * NPAD + k] : NEGV;
    q = fmaxf(q, cand);
  }
  query[(long)b * NPAD + k] = q;
}

__global__ void alpha_kernel(const float* __restrict__ v, const float* __restrict__ query,
                             const int* __restrict__ text, float* __restrict__ alpha) {
  int i = blockIdx.x * 256 + threadIdx.x;
  if (i >= BN * SN) return;
  int b = i >> 7;
  float s = 0.f;
  for (int k = 0; k < 2 * HN; ++k) s += v[(long)i * NPAD + k] * query[(long)b * NPAD + k];
  alpha[i] = (text[i] != 0) ? s : NEGV;
}

__global__ void alpha_softmax_kernel(float* __restrict__ alpha) {
  __shared__ float red[SN];
  int b = blockIdx.x, s = threadIdx.x;
  float x = alpha[b * SN + s];
  red[s] = x;
  __syncthreads();
  for (int off = SN / 2; off > 0; off >>= 1) {
    if (s < off) red[s] = fmaxf(red[s], red[s + off]);
    __syncthreads();
  }
  float mx = red[0];
  __syncthreads();
  float ex = __expf(x - mx);
  red[s] = ex;
  __syncthreads();
  for (int off = SN / 2; off > 0; off >>= 1) {
    if (s < off) red[s] += red[s + off];
    __syncthreads();
  }
  alpha[b * SN + s] = ex / red[0];
}

__global__ void zfinal_kernel(const float* __restrict__ alpha, const float* __restrict__ v,
                              float* __restrict__ z) {
  int i = blockIdx.x * 256 + threadIdx.x;
  if (i >= BN * 2 * HN) return;
  int b = i / (2 * HN), k = i % (2 * HN);
  float s = 0.f;
  for (int t = 0; t < SN; ++t) s += alpha[b * SN + t] * v[((long)b * SN + t) * NPAD + k];
  z[(long)b * NPAD + k] = s;
}

__global__ void fc_kernel(const float* __restrict__ z, const float* __restrict__ W,
                          const float* __restrict__ bias, float* __restrict__ out) {
  int i = blockIdx.x * 64 + threadIdx.x;
  if (i >= BN * 3) return;
  int b = i / 3, o = i % 3;
  float s = bias[o];
  for (int k = 0; k < 2 * HN; ++k) s += z[(long)b * NPAD + k] * W[o * 2 * HN + k];
  out[i] = s;
}

// ---------------------------------------------------------------------------
static inline char* ws_take(char*& p, size_t bytes) {
  char* r = p;
  p += (bytes + 255) & ~(size_t)255;
  return r;
}

extern "C" void kernel_launch(void* const* d_in, const int* in_sizes, int n_in,
                              void* d_out, int out_size, void* d_ws, size_t ws_size,
                              hipStream_t stream) {
  (void)in_sizes; (void)n_in; (void)out_size; (void)ws_size;
  const int*   text   = (const int*)d_in[0];
  const int*   asp    = (const int*)d_in[1];
  const float* poswt  = (const float*)d_in[2];
  const float* emb    = (const float*)d_in[3];
  const float* l1Wih  = (const float*)d_in[4];
  const float* l1Whh  = (const float*)d_in[5];
  const float* l1bih  = (const float*)d_in[6];
  const float* l1bhh  = (const float*)d_in[7];
  const float* l2Wih  = (const float*)d_in[8];
  const float* l2Whh  = (const float*)d_in[9];
  const float* l2bih  = (const float*)d_in[10];
  const float* l2bhh  = (const float*)d_in[11];
  const float* lin1W  = (const float*)d_in[12];
  const float* lin1b  = (const float*)d_in[13];
  const float* lin2W  = (const float*)d_in[14];
  const float* lin2b  = (const float*)d_in[15];
  const float* fcW    = (const float*)d_in[16];
  const float* fcb    = (const float*)d_in[17];
  float* out = (float*)d_out;

  char* p = (char*)d_ws;
  _Float16* wf16   = (_Float16*)ws_take(p, (size_t)BN * SN * KP * 2);
  _Float16* af16   = (_Float16*)ws_take(p, (size_t)BN * AN * KP * 2);
  _Float16* W1ihT  = (_Float16*)ws_take(p, (size_t)2 * KP * GCOL * 2);
  _Float16* W1hhT  = (_Float16*)ws_take(p, (size_t)2 * KP * GCOL * 2);
  _Float16* W2ihT  = (_Float16*)ws_take(p, (size_t)2 * KP * GCOL * 2);
  _Float16* W2hhT  = (_Float16*)ws_take(p, (size_t)2 * KP * GCOL * 2);
  float*    bsum1  = (float*)ws_take(p, (size_t)2 * GCOL * 4);
  float*    bsum2  = (float*)ws_take(p, (size_t)2 * GCOL * 4);
  _Float16* lin1WT = (_Float16*)ws_take(p, (size_t)2 * GCOL * NPAD * 2);
  float*    lin1bp = (float*)ws_take(p, (size_t)2 * NPAD * 4);
  _Float16* gpre1  = (_Float16*)ws_take(p, (size_t)2 * BN * SN * GCOL * 2);
  _Float16* gpre2  = (_Float16*)ws_take(p, (size_t)2 * BN * AN * GCOL * 2);
  _Float16* h1     = (_Float16*)ws_take(p, (size_t)2 * 2 * BN * KP * 2);  // parity x dir
  float*    c1     = (float*)ws_take(p, (size_t)2 * BN * HPAD * 4);
  _Float16* h2     = (_Float16*)ws_take(p, (size_t)2 * 2 * BN * KP * 2);
  float*    c2     = (float*)ws_take(p, (size_t)2 * BN * HPAD * 4);
  float*    v      = (float*)ws_take(p, (size_t)BN * SN * NPAD * 4);
  float*    e      = (float*)ws_take(p, (size_t)BN * AN * NPAD * 4);
  _Float16* v16    = (_Float16*)ws_take(p, (size_t)BN * SN * NPAD * 2);
  _Float16* e16    = (_Float16*)ws_take(p, (size_t)BN * 64 * NPAD * 2);
  _Float16* eT16   = (_Float16*)ws_take(p, (size_t)BN * NPAD * 64 * 2);
  float*    a32    = (float*)ws_take(p, (size_t)BN * SN * 64 * 4);
  _Float16* a16    = (_Float16*)ws_take(p, (size_t)BN * SN * 64 * 2);
  float*    mid    = (float*)ws_take(p, (size_t)BN * SN * NPAD * 4);
  _Float16* cc     = (_Float16*)ws_take(p, (size_t)BN * SN * GCOL * 2);
  float*    l1out  = (float*)ws_take(p, (size_t)BN * SN * NPAD * 4);
  float*    tg     = (float*)ws_take(p, (size_t)BN * SN * 4);
  float*    query  = (float*)ws_take(p, (size_t)BN * NPAD * 4);
  float*    alpha  = (float*)ws_take(p, (size_t)BN * SN * 4);
  float*    z      = (float*)ws_take(p, (size_t)BN * NPAD * 4);
  int*      lens   = (int*)ws_take(p, (size_t)2 * BN * 4);

  auto zero = [&](void* ptr, size_t words) {
    int blocks = (int)((words + 255) / 256);
    zero_kernel<<<blocks, 256, 0, stream>>>((unsigned int*)ptr, (long)words);
  };
  // state buffers must be zero every call (harness does not re-poison/restore)
  zero(v,  (size_t)BN * SN * NPAD);
  zero(e,  (size_t)BN * AN * NPAD);
  zero(h1, (size_t)2 * 2 * BN * KP / 2);  // f16 -> words
  zero(c1, (size_t)2 * BN * HPAD);
  zero(h2, (size_t)2 * 2 * BN * KP / 2);
  zero(c2, (size_t)2 * BN * HPAD);

  lens_kernel<<<1, 64, 0, stream>>>(text, asp, lens);

  {
    long tot = 2L * KP * GCOL;
    int blk = (int)((tot + 255) / 256);
    conv_wT_kernel<<<blk, 256, 0, stream>>>(l1Wih, W1ihT);
    conv_wT_kernel<<<blk, 256, 0, stream>>>(l1Whh, W1hhT);
    conv_wT_kernel<<<blk, 256, 0, stream>>>(l2Wih, W2ihT);
    conv_wT_kernel<<<blk, 256, 0, stream>>>(l2Whh, W2hhT);
  }
  bsum_kernel<<<(2 * GCOL + 255) / 256, 256, 0, stream>>>(l1bih, l1bhh, bsum1);
  bsum_kernel<<<(2 * GCOL + 255) / 256, 256, 0, stream>>>(l2bih, l2bhh, bsum2);
  conv_lin1_kernel<<<(int)((2L * GCOL * NPAD + 255) / 256), 256, 0, stream>>>(lin1W, lin1WT);
  pad_lin1b_kernel<<<(2 * NPAD + 255) / 256, 256, 0, stream>>>(lin1b, lin1bp);

  gather_kernel<<<(int)(((long)BN * SN * KP + 255) / 256), 256, 0, stream>>>(text, emb, wf16, (long)BN * SN);
  gather_kernel<<<(int)(((long)BN * AN * KP + 255) / 256), 256, 0, stream>>>(asp, emb, af16, (long)BN * AN);

  // gpre = x @ Wih^T + (bih+bhh)   [WMMA, both directions]
  for (int d = 0; d < 2; ++d) {
    gemm_wmma_kernel<true><<<dim3(GCOL / 64, (BN * SN) / 128, 1), 256, 0, stream>>>(
        wf16, KP, 0, W1ihT + (long)d * KP * GCOL, GCOL, 0,
        (void*)(gpre1 + (long)d * BN * SN * GCOL), GCOL, 0, KP, bsum1 + d * GCOL);
    gemm_wmma_kernel<true><<<dim3(GCOL / 64, (BN * AN) / 128, 1), 256, 0, stream>>>(
        af16, KP, 0, W2ihT + (long)d * KP * GCOL, GCOL, 0,
        (void*)(gpre2 + (long)d * BN * AN * GCOL), GCOL, 0, KP, bsum2 + d * GCOL);
  }

  // layer-1 BiLSTM over text (128 sequential steps, graph-captured)
  const long hpar = 2L * BN * KP;
  for (int t = 0; t < SN; ++t) {
    lstm_step_kernel<<<dim3(19, 2), 256, 0, stream>>>(
        gpre1, W1hhT, h1 + (t & 1) * hpar, h1 + ((t + 1) & 1) * hpar, c1, v, lens, t, SN);
  }
  // layer-2 BiLSTM over aspect (8 steps)
  for (int t = 0; t < AN; ++t) {
    lstm_step_kernel<<<dim3(19, 2), 256, 0, stream>>>(
        gpre2, W2hhT, h2 + (t & 1) * hpar, h2 + ((t + 1) & 1) * hpar, c2, e, lens + BN, t, AN);
  }

  cvt_f32_f16_kernel<<<(int)(((long)BN * SN * NPAD + 255) / 256), 256, 0, stream>>>(v, v16, (long)BN * SN * NPAD);
  build_e16_kernel<<<(int)(((long)BN * 64 * NPAD + 255) / 256), 256, 0, stream>>>(e, e16, eT16);

  for (int it = 0; it < 2; ++it) {
    // a = v @ e^T  (per-batch WMMA)
    gemm_wmma_kernel<false><<<dim3(1, SN / 128, BN), 256, 0, stream>>>(
        v16, NPAD, (long)SN * NPAD, eT16, 64, (long)NPAD * 64,
        (void*)a32, 64, (long)SN * 64, NPAD, nullptr);
    attn_softmax_kernel<<<(BN * SN + 255) / 256, 256, 0, stream>>>(a32, text, asp, a16);
    // mid = softmax(a) @ e
    gemm_wmma_kernel<false><<<dim3(NPAD / 64, SN / 128, BN), 256, 0, stream>>>(
        a16, 64, (long)SN * 64, e16, NPAD, (long)64 * NPAD,
        (void*)mid, NPAD, (long)SN * NPAD, 64, nullptr);
    concat_kernel<<<(int)(((long)BN * SN * GCOL + 255) / 256), 256, 0, stream>>>(mid, v, cc);
    // lin1: [8192,1216] x [1216,640]
    gemm_wmma_kernel<false><<<dim3(NPAD / 64, (BN * SN) / 128, 1), 256, 0, stream>>>(
        cc, GCOL, 0, lin1WT + (long)it * GCOL * NPAD, NPAD, 0,
        (void*)l1out, NPAD, 0, GCOL, lin1bp + it * NPAD);
    tgate_kernel<<<(BN * SN + 255) / 256, 256, 0, stream>>>(v, lin2W + (long)it * 2 * HN, lin2b + it, tg);
    vupdate_kernel<<<(int)(((long)BN * SN * NPAD + 255) / 256), 256, 0, stream>>>(v, v16, l1out, tg, poswt);
  }

  query_kernel<<<(BN * 2 * HN + 255) / 256, 256, 0, stream>>>(e, asp, query);
  alpha_kernel<<<(BN * SN + 255) / 256, 256, 0, stream>>>(v, query, text, alpha);
  alpha_softmax_kernel<<<BN, SN, 0, stream>>>(alpha);
  zfinal_kernel<<<(BN * 2 * HN + 255) / 256, 256, 0, stream>>>(alpha, v, z);
  fc_kernel<<<(BN * 3 + 63) / 64, 64, 0, stream>>>(z, fcW, fcb, out);
}
